// GCN_38302518346058
// MI455X (gfx1250) — matrix operand order repaired and checked by reading, not compiled, not run on hardware
//
#include <hip/hip_runtime.h>

// ---------------------------------------------------------------------------
// GCN (2x GCNConv + linear) for MI455X / gfx1250, wave32, WMMA bf16.
// ---------------------------------------------------------------------------

typedef __attribute__((ext_vector_type(16))) __bf16 v16bf;
typedef __attribute__((ext_vector_type(8)))  float  v8f;

union FragBF { v16bf v; unsigned int d[8]; uint4 q[2]; };
union FragF  { v8f   v; float f[8]; };

__device__ __forceinline__ unsigned short f2bf(float x) {
    unsigned int u = __builtin_bit_cast(unsigned int, x);
    // round-to-nearest-even truncation to bf16
    return (unsigned short)((u + 0x7FFFu + ((u >> 16) & 1u)) >> 16);
}

// ---------------------------------------------------------------------------
// Small utility kernels
// ---------------------------------------------------------------------------
__global__ void init_ones_kernel(float* __restrict__ p, int n) {
    int i = blockIdx.x * blockDim.x + threadIdx.x;
    if (i < n) p[i] = 1.0f;
}

__global__ void edge_deg_kernel(const long long* __restrict__ dst,
                                float* __restrict__ deg, int nE) {
    int e = blockIdx.x * blockDim.x + threadIdx.x;
    if (e < nE) {
        int d = (int)dst[e];
        __hip_atomic_fetch_add(&deg[d], 1.0f, __ATOMIC_RELAXED, __HIP_MEMORY_SCOPE_AGENT);
    }
}

__global__ void rsqrt_kernel(float* __restrict__ p, int n) {
    int i = blockIdx.x * blockDim.x + threadIdx.x;
    if (i < n) p[i] = rsqrtf(p[i]);
}

// f32 -> bf16, 4 elements / thread (float4 in, ushort4 out), optional ReLU.
// n must be a multiple of 4 (always true here: N*128 / N*256).
__global__ void f32_to_bf16_kernel(const float* __restrict__ in,
                                   unsigned short* __restrict__ out,
                                   long long n4, int relu) {
    long long i = (long long)blockIdx.x * blockDim.x + threadIdx.x;
    if (i >= n4) return;
    float4 v = ((const float4*)in)[i];
    if (relu) {
        v.x = fmaxf(v.x, 0.0f); v.y = fmaxf(v.y, 0.0f);
        v.z = fmaxf(v.z, 0.0f); v.w = fmaxf(v.w, 0.0f);
    }
    ushort4 o;
    o.x = f2bf(v.x); o.y = f2bf(v.y); o.z = f2bf(v.z); o.w = f2bf(v.w);
    ((ushort4*)out)[i] = o;
}

// W [K,N] f32 row-major  ->  BT [N,K] bf16 row-major (transposed weights)
__global__ void weight_to_bf16T_kernel(const float* __restrict__ W,
                                       unsigned short* __restrict__ BT,
                                       int K, int N) {
    int i = blockIdx.x * blockDim.x + threadIdx.x;
    if (i >= K * N) return;
    int k = i / N, n = i % N;
    BT[(long long)n * K + k] = f2bf(W[i]);
}

// ---------------------------------------------------------------------------
// GEMM: C[M,256] = A[M,K](bf16) @ B[K,256], B stored transposed BT[256,K](bf16).
// Each wave computes a 16x64 strip: 4 accumulators share one A-fragment.
// K is a template constant (128/256) so the K-loop fully unrolls and all
// load offsets fold into immediate ioffsets (b128 loads, no 64-bit addr math).
// ---------------------------------------------------------------------------
template <int K>
__global__ void gemm_bf16_wmma_kernel(const unsigned short* __restrict__ A,
                                      const unsigned short* __restrict__ BT,
                                      float* __restrict__ C,
                                      int M,
                                      const float* __restrict__ bias,
                                      int relu) {
    constexpr int N = 256;
    const int lane = threadIdx.x & 31;
    const int wave = threadIdx.x >> 5;
    const int ngrp = blockIdx.x;                 // group of 4 n-tiles (64 cols)
    const int mtile = blockIdx.y * 8 + wave;     // wave-uniform
    if (mtile >= (M >> 4)) return;               // whole wave exits: EXEC all-1s

    const int l16  = lane & 15;
    const int half = lane >> 4;

    // A row for this lane; the two 16B chunks of the A-frag are at +0 and +32B.
    const unsigned short* aptr =
        A + (long long)((mtile << 4) + l16) * K + half * 8;
    // B column (= BT row) base for tile t is +t*16 rows.
    const unsigned short* bptr =
        BT + (long long)((ngrp << 6) + l16) * K + half * 8;

    FragF acc[4];
#pragma unroll
    for (int t = 0; t < 4; ++t)
#pragma unroll
        for (int r = 0; r < 8; ++r) acc[t].f[r] = 0.0f;

#pragma unroll
    for (int kb = 0; kb < K; kb += 32) {
        FragBF a;
        const uint4* ap = (const uint4*)(aptr + kb);
        a.q[0] = ap[0];      // K pairs {0..7}  (+half*8)
        a.q[1] = ap[2];      // K pairs {16..23}(+half*8)  (+32 bytes)

#pragma unroll
        for (int t = 0; t < 4; ++t) {
            FragBF b;
            const uint4* bp = (const uint4*)(bptr + (long long)t * 16 * K + kb);
            b.q[0] = bp[0];
            b.q[1] = bp[2];
            acc[t].v = __builtin_amdgcn_wmma_f32_16x16x32_bf16(
                /*neg_a=*/false, a.v, /*neg_b=*/false, b.v,
                /*c_mod=*/(short)0, acc[t].v,
                /*reuse_a=*/false, /*reuse_b=*/false);
        }
    }

    // C store: acc[t].f[r] -> row = mtile*16 + half*8 + r, col = ngrp*64+t*16+l16
    const int row0 = (mtile << 4) + half * 8;
#pragma unroll
    for (int t = 0; t < 4; ++t) {
        const int n = (ngrp << 6) + (t << 4) + l16;
        const float badd = bias ? bias[n] : 0.0f;
        float* cp = C + (long long)row0 * N + n;
#pragma unroll
        for (int r = 0; r < 8; ++r) {
            float val = acc[t].f[r] + badd;
            if (relu) val = fmaxf(val, 0.0f);
            cp[r * N] = val;
        }
    }
}

// ---------------------------------------------------------------------------
// Aggregation init: out[i,:] = h[i,:] * dinv[i]^2 + bias  (self-loop + bias)
// float4 vectorized; C == 256.
// ---------------------------------------------------------------------------
__global__ void self_init_kernel(const float* __restrict__ h,
                                 const float* __restrict__ dinv,
                                 const float* __restrict__ bias,
                                 float* __restrict__ out,
                                 int nNodes) {
    long long i = (long long)blockIdx.x * blockDim.x + threadIdx.x; // float4 idx
    long long total = (long long)nNodes * 64;                       // 256/4
    if (i >= total) return;
    int node = (int)(i >> 6);
    int c4   = (int)(i & 63);
    float w = dinv[node];
    float ww = w * w;
    float4 v = ((const float4*)h)[i];
    float4 b = ((const float4*)bias)[c4];
    float4 o;
    o.x = v.x * ww + b.x; o.y = v.y * ww + b.y;
    o.z = v.z * ww + b.z; o.w = v.w * ww + b.w;
    ((float4*)out)[i] = o;
}

// ---------------------------------------------------------------------------
// Edge scatter: out[dst,:] += dinv[src]*dinv[dst] * h[src,:]
// One wave per edge, 8 channels per lane (C == 256). f32 HW atomics; the
// 51 MB activation matrix is L2-resident (192 MB L2) so this is L2-bound.
// ---------------------------------------------------------------------------
__global__ void edge_scatter_kernel(const float* __restrict__ h,
                                    const long long* __restrict__ src,
                                    const long long* __restrict__ dst,
                                    const float* __restrict__ dinv,
                                    float* __restrict__ out, int nE) {
    int e = blockIdx.x * 8 + (threadIdx.x >> 5);
    if (e >= nE) return;
    int lane = threadIdx.x & 31;
    int s = (int)src[e];
    int d = (int)dst[e];
    float w = dinv[s] * dinv[d];

    const float4* hp = (const float4*)(h + (long long)s * 256);
    float* op = out + (long long)d * 256;

    float4 v0 = hp[lane * 2];
    float4 v1 = hp[lane * 2 + 1];
    int c = lane * 8;
    __hip_atomic_fetch_add(op + c + 0, v0.x * w, __ATOMIC_RELAXED, __HIP_MEMORY_SCOPE_AGENT);
    __hip_atomic_fetch_add(op + c + 1, v0.y * w, __ATOMIC_RELAXED, __HIP_MEMORY_SCOPE_AGENT);
    __hip_atomic_fetch_add(op + c + 2, v0.z * w, __ATOMIC_RELAXED, __HIP_MEMORY_SCOPE_AGENT);
    __hip_atomic_fetch_add(op + c + 3, v0.w * w, __ATOMIC_RELAXED, __HIP_MEMORY_SCOPE_AGENT);
    __hip_atomic_fetch_add(op + c + 4, v1.x * w, __ATOMIC_RELAXED, __HIP_MEMORY_SCOPE_AGENT);
    __hip_atomic_fetch_add(op + c + 5, v1.y * w, __ATOMIC_RELAXED, __HIP_MEMORY_SCOPE_AGENT);
    __hip_atomic_fetch_add(op + c + 6, v1.z * w, __ATOMIC_RELAXED, __HIP_MEMORY_SCOPE_AGENT);
    __hip_atomic_fetch_add(op + c + 7, v1.w * w, __ATOMIC_RELAXED, __HIP_MEMORY_SCOPE_AGENT);
}

// ---------------------------------------------------------------------------
// Orchestration
// ---------------------------------------------------------------------------
extern "C" void kernel_launch(void* const* d_in, const int* in_sizes, int n_in,
                              void* d_out, int out_size, void* d_ws, size_t ws_size,
                              hipStream_t stream) {
    const int IN_C = 128, HID = 256, OUT = 256, EMB = 256;
    const int N = in_sizes[0] / IN_C;        // 50000
    const int E = in_sizes[7] / 2;           // 800000

    const float*     x    = (const float*)d_in[0];
    const float*     W1   = (const float*)d_in[1];
    const float*     b1   = (const float*)d_in[2];
    const float*     W2   = (const float*)d_in[3];
    const float*     b2   = (const float*)d_in[4];
    const float*     Wlin = (const float*)d_in[5];
    const float*     blin = (const float*)d_in[6];
    const long long* ei   = (const long long*)d_in[7];
    const long long* srcI = ei;              // edge_index[0]
    const long long* dstI = ei + E;          // edge_index[1]
    float* out = (float*)d_out;

    // workspace layout
    char* ws = (char*)d_ws;
    size_t off = 0;
    auto walloc = [&](size_t bytes) -> void* {
        void* p = ws + off;
        off = (off + bytes + 255) & ~(size_t)255;
        return p;
    };
    float*          dinv = (float*)walloc((size_t)N * 4);
    float*          hbuf = (float*)walloc((size_t)N * 256 * 4);   // GEMM outputs
    float*          agg  = (float*)walloc((size_t)N * 256 * 4);   // aggregation outputs
    unsigned short* abf  = (unsigned short*)walloc((size_t)N * 256 * 2);
    unsigned short* xbf  = (unsigned short*)walloc((size_t)N * IN_C * 2);
    unsigned short* W1T  = (unsigned short*)walloc((size_t)IN_C * HID * 2);
    unsigned short* W2T  = (unsigned short*)walloc((size_t)HID * OUT * 2);
    unsigned short* WlT  = (unsigned short*)walloc((size_t)OUT * EMB * 2);
    (void)ws_size;

    const int TB = 256;
    // --- degree / normalization ---
    init_ones_kernel<<<(N + TB - 1) / TB, TB, 0, stream>>>(dinv, N);
    edge_deg_kernel<<<(E + TB - 1) / TB, TB, 0, stream>>>(dstI, dinv, E);
    rsqrt_kernel<<<(N + TB - 1) / TB, TB, 0, stream>>>(dinv, N);

    // --- weight conversions (transposed bf16) ---
    weight_to_bf16T_kernel<<<(IN_C * HID + TB - 1) / TB, TB, 0, stream>>>(W1, W1T, IN_C, HID);
    weight_to_bf16T_kernel<<<(HID * OUT + TB - 1) / TB, TB, 0, stream>>>(W2, W2T, HID, OUT);
    weight_to_bf16T_kernel<<<(OUT * EMB + TB - 1) / TB, TB, 0, stream>>>(Wlin, WlT, OUT, EMB);

    // --- x -> bf16 ---
    long long nx4 = (long long)N * IN_C / 4;
    f32_to_bf16_kernel<<<(unsigned)((nx4 + TB - 1) / TB), TB, 0, stream>>>(x, xbf, nx4, 0);

    long long nh4 = (long long)N * 64;               // N*256/4
    const int Mtiles = N >> 4;                       // 3125
    dim3 gGemm(256 / 64, (Mtiles + 7) / 8);          // 4 x 391
    dim3 gEdges((E + 7) / 8);                        // 1 edge / wave, 8 waves / block
    unsigned gElem = (unsigned)((nh4 + TB - 1) / TB);

    // ================= Layer 1: h1 = x @ W1 ; agg + b1 ; relu =================
    gemm_bf16_wmma_kernel<128><<<gGemm, TB, 0, stream>>>(xbf, W1T, hbuf, N, nullptr, 0);
    self_init_kernel<<<gElem, TB, 0, stream>>>(hbuf, dinv, b1, agg, N);
    edge_scatter_kernel<<<gEdges, TB, 0, stream>>>(hbuf, srcI, dstI, dinv, agg, E);
    f32_to_bf16_kernel<<<gElem, TB, 0, stream>>>(agg, abf, nh4, /*relu=*/1);

    // ================= Layer 2: h2 = a1 @ W2 ; agg + b2 (no relu) =============
    gemm_bf16_wmma_kernel<256><<<gGemm, TB, 0, stream>>>(abf, W2T, hbuf, N, nullptr, 0);
    self_init_kernel<<<gElem, TB, 0, stream>>>(hbuf, dinv, b2, agg, N);
    edge_scatter_kernel<<<gEdges, TB, 0, stream>>>(hbuf, srcI, dstI, dinv, agg, E);
    f32_to_bf16_kernel<<<gElem, TB, 0, stream>>>(agg, abf, nh4, /*relu=*/0);

    // ================= Head: out = relu(a2 @ Wlin + blin) =====================
    gemm_bf16_wmma_kernel<256><<<gGemm, TB, 0, stream>>>(abf, WlT, out, N, blin, /*relu=*/1);
}
